// AttGNNClassifier_46858093199923
// MI455X (gfx1250) — matrix operand order repaired and checked by reading, not compiled, not run on hardware
//
#include <hip/hip_runtime.h>
#include <hip/hip_bf16.h>

// ---------------- problem constants (match reference) ----------------
#define NN 50000
#define EE 400000
#define F_IN 128
#define HH 3
#define DD 64
#define GG 64
#define PP 64
#define HD 192            // H*D
#define NTC 12            // number of 16-wide column tiles in HD
#define NEG_GAT 0.2f
#define NEG_MLP 0.01f

typedef __attribute__((ext_vector_type(16))) __bf16 v16bf;
typedef __attribute__((ext_vector_type(8)))  float  v8f;

#if __has_builtin(__builtin_amdgcn_global_load_async_to_lds_b32) && \
    __has_builtin(__builtin_amdgcn_s_wait_asynccnt)
#define HAVE_ASYNC_LDS 1
typedef __attribute__((address_space(1))) int as1_int;
typedef __attribute__((address_space(3))) int as3_int;
#else
#define HAVE_ASYNC_LDS 0
#endif

// ---------------- small helpers ----------------
__device__ __forceinline__ unsigned int f32key(float f) {
  unsigned int u = __float_as_uint(f);
  return (u & 0x80000000u) ? ~u : (u | 0x80000000u);   // order-preserving map
}
__device__ __forceinline__ float keyf32(unsigned int k) {
  unsigned int u = (k & 0x80000000u) ? (k ^ 0x80000000u) : ~k;
  return __uint_as_float(u);
}
__device__ __forceinline__ unsigned int f2bf(float f) {
  unsigned int u = __float_as_uint(f);
  return (u + 0x7FFFu + ((u >> 16) & 1u)) >> 16;
}

// fp32 -> bf16 (round to nearest even), flat array
__global__ void cvt_bf16_k(const float* __restrict__ x, unsigned short* __restrict__ y, int n) {
  int i = blockIdx.x * blockDim.x + threadIdx.x;
  if (i >= n) return;
  y[i] = (unsigned short)f2bf(x[i]);
}

// fp32 W[K x 192] -> tile-packed bf16. Tile (tk, tc) = 32x16 block at
// k0 = tk*32, col0 = tc*16; occupies 256 contiguous dwords. Dword (n, kp)
// holds bf16 pair (k = 2*kp, 2*kp+1) of column col0+n -> a lane's B fragment
// is two 4-dword runs at kp = 4*half and 4*half+8, and staging the tile is a contiguous async copy.
__global__ void cvt_w_pack_k(const float* __restrict__ W, unsigned int* __restrict__ Wpk, int K) {
  int t = blockIdx.x * blockDim.x + threadIdx.x;
  const int total = (K >> 5) * NTC * 256;
  if (t >= total) return;
  const int tile = t >> 8;
  const int idx  = t & 255;
  const int tk = tile / NTC, tc = tile % NTC;
  const int n = idx >> 4, kp = idx & 15;
  const int k   = tk * 32 + 2 * kp;
  const int col = tc * 16 + n;
  const unsigned int u0 = f2bf(W[(size_t)k * HD + col]);
  const unsigned int u1 = f2bf(W[(size_t)(k + 1) * HD + col]);
  Wpk[t] = (u1 << 16) | u0;
}

__global__ void fill_f32_k(float* __restrict__ p, float v, int n) {
  int i = blockIdx.x * blockDim.x + threadIdx.x;
  if (i < n) p[i] = v;
}
__global__ void fill_u32_k(unsigned int* __restrict__ p, unsigned int v, int n) {
  int i = blockIdx.x * blockDim.x + threadIdx.x;
  if (i < n) p[i] = v;
}

// ---------------- fused dual WMMA GEMM: fs = Xb@Ws, fd = Xb@Wd ----------------
// Xb: [Nrows x K] bf16 row-major. WsPk/WdPk: tile-packed (see cvt_w_pack_k).
// fs/fd: [Nrows x 192] fp32. Block = 8 waves; block tile = 128 rows x 16 cols.
// W tiles staged into LDS with gfx1250 async global->LDS (ASYNCcnt); A
// fragments are two b128 global loads per lane; B fragments two b128 ds loads.
__global__ __launch_bounds__(256) void gemm_dual_wmma_k(
    const unsigned short* __restrict__ Xb,
    const unsigned int* __restrict__ WsPk,
    const unsigned int* __restrict__ WdPk,
    float* __restrict__ fs, float* __restrict__ fd,
    int Nrows, int K)
{
  __shared__ unsigned int ldsS[256];   // one packed 32x16 tile each
  __shared__ unsigned int ldsD[256];

  const int tid  = threadIdx.x;
  const int wave = tid >> 5;
  const int lane = tid & 31;
  const int half = lane >> 4;      // which 16-lane half
  const int nl   = lane & 15;

  int rowBase = blockIdx.x * 128 + wave * 16;
  if (rowBase + 16 > Nrows) rowBase = Nrows - 16;   // Nrows % 16 == 0 -> safe duplicate
  const int col0 = blockIdx.y * 16;

  v8f accS = {};
  v8f accD = {};
  const uint4* X4 = (const uint4*)Xb;
  const int row = rowBase + nl;

  for (int k0 = 0; k0 < K; k0 += 32) {
    // stage both packed W tiles: contiguous 1KB per matrix
    {
      const size_t tileBase = ((size_t)(k0 >> 5) * NTC + blockIdx.y) * 256;
      const unsigned int* gS = WsPk + tileBase + tid;
      const unsigned int* gD = WdPk + tileBase + tid;
#if HAVE_ASYNC_LDS
      __builtin_amdgcn_global_load_async_to_lds_b32(
          (as1_int*)gS, (as3_int*)&ldsS[tid], 0, 0);
      __builtin_amdgcn_global_load_async_to_lds_b32(
          (as1_int*)gD, (as3_int*)&ldsD[tid], 0, 0);
      __builtin_amdgcn_s_wait_asynccnt(0);
#else
      ldsS[tid] = *gS;
      ldsD[tid] = *gD;
#endif
    }
    __syncthreads();

    // A fragment: 16x32 bf16 per ISA layout. VGPRs 0-3 = 16 contiguous bytes
    // at K = k0 + 8*half, VGPRs 4-7 = 16 contiguous bytes at K + 16.
    union { v16bf v; uint4 q[2]; } A;
    const size_t aBase = ((size_t)row * K + k0 + 8 * half) >> 3;  // uint4 index
    A.q[0] = X4[aBase];
    A.q[1] = X4[aBase + 2];

    // B fragments: lane covers col nl, 4 dwords at K = 8*half and 4 dwords at K = 16 + 8*half
    union { v16bf v; uint4 q[2]; } BS, BD;
    const uint4* pS = (const uint4*)&ldsS[nl * 16 + 4 * half];
    const uint4* pD = (const uint4*)&ldsD[nl * 16 + 4 * half];
    BS.q[0] = pS[0]; BS.q[1] = pS[2];
    BD.q[0] = pD[0]; BD.q[1] = pD[2];

    accS = __builtin_amdgcn_wmma_f32_16x16x32_bf16(false, A.v, false, BS.v,
                                                   (short)0, accS, false, false);
    accD = __builtin_amdgcn_wmma_f32_16x16x32_bf16(false, A.v, false, BD.v,
                                                   (short)0, accD, false, false);
    __syncthreads();

    if (k0 + 32 < K) __builtin_prefetch(&Xb[(size_t)row * K + k0 + 32], 0, 0);
  }

  // C/D layout: VGPR r -> row r (lanes 0-15) / r+8 (lanes 16-31), col = lane%16
#pragma unroll
  for (int r = 0; r < 8; ++r) {
    const int orow = rowBase + r + 8 * half;
    fs[(size_t)orow * HD + col0 + nl] = accS[r];
    fd[(size_t)orow * HD + col0 + nl] = accD[r];
  }
}

// ---------------- GATv2 edge kernels ----------------
// one thread per (edge, head)
__global__ void edge_logits_k(const int* __restrict__ src, const int* __restrict__ dst,
                              const float* __restrict__ fs, const float* __restrict__ fd,
                              const float* __restrict__ attn,
                              float* __restrict__ logits, unsigned int* __restrict__ menc)
{
  int t = blockIdx.x * blockDim.x + threadIdx.x;
  if (t >= EE * HH) return;
  const int e = t / HH, hh = t % HH;
  const float* ps = fs + (size_t)src[e] * HD + hh * DD;
  const float* pf = fd + (size_t)dst[e] * HD + hh * DD;
  const float* pa = attn + hh * DD;
  float s = 0.f;
#pragma unroll 8
  for (int d = 0; d < DD; ++d) {
    float v = ps[d] + pf[d];
    v = v > 0.f ? v : NEG_GAT * v;
    s += pa[d] * v;
  }
  logits[t] = s;
  atomicMax(&menc[dst[e] * HH + hh], f32key(s));
}

__global__ void mfix_k(const unsigned int* __restrict__ menc, float* __restrict__ mf, int n) {
  int i = blockIdx.x * blockDim.x + threadIdx.x;
  if (i >= n) return;
  unsigned int k = menc[i];
  mf[i] = (k == 0u) ? 0.f : keyf32(k);     // no in-edges -> 0 (matches isfinite fixup)
}

__global__ void edge_exp_k(const int* __restrict__ dst,
                           const float* __restrict__ logits, const float* __restrict__ mf,
                           float* __restrict__ ex, float* __restrict__ denom)
{
  int t = blockIdx.x * blockDim.x + threadIdx.x;
  if (t >= EE * HH) return;
  const int e = t / HH, hh = t % HH;
  float x = __expf(logits[t] - mf[dst[e] * HH + hh]);
  ex[t] = x;
  atomicAdd(&denom[dst[e] * HH + hh], x);
}

__global__ void edge_agg_k(const int* __restrict__ src, const int* __restrict__ dst,
                           const float* __restrict__ fs,
                           const float* __restrict__ ex, const float* __restrict__ denom,
                           float* __restrict__ acc)
{
  int t = blockIdx.x * blockDim.x + threadIdx.x;
  if (t >= EE * HH) return;
  const int e = t / HH, hh = t % HH;
  const float alpha = ex[t] / fmaxf(denom[dst[e] * HH + hh], 1e-9f);
  const float* ps = fs + (size_t)src[e] * HD + hh * DD;
  float* po = acc + (size_t)dst[e] * HD + hh * DD;
#pragma unroll 4
  for (int d = 0; d < DD; ++d) atomicAdd(&po[d], alpha * ps[d]);
}

__global__ void add_bias_k(float* __restrict__ acc, const float* __restrict__ bias, int n) {
  int i = blockIdx.x * blockDim.x + threadIdx.x;
  if (i < n) acc[i] += bias[i % HD];
}

// ---------------- readout ----------------
__global__ void hmean_k(const float* __restrict__ acc, const int* __restrict__ gid,
                        float* __restrict__ gsum, float* __restrict__ gcnt)
{
  int i = blockIdx.x * blockDim.x + threadIdx.x;
  if (i >= NN * DD) return;
  const int n = i / DD, d = i % DD;
  const float v = (acc[(size_t)n * HD + d] + acc[(size_t)n * HD + DD + d] +
                   acc[(size_t)n * HD + 2 * DD + d]) * (1.f / 3.f);
  const int g = gid[n];
  atomicAdd(&gsum[g * DD + d], v);
  if (d == 0) atomicAdd(&gcnt[g], 1.f);
}

__global__ void gfinal_k(const float* __restrict__ gsum, const float* __restrict__ gcnt,
                         float* __restrict__ xcat)
{
  int i = blockIdx.x * blockDim.x + threadIdx.x;
  if (i >= GG * DD) return;
  const int g = i / DD, d = i % DD;
  xcat[g * (DD + PP) + d] = gsum[i] / fmaxf(gcnt[g], 1.f);
}

// ---------------- generic small dense layer (head / pattern branch) ----------------
__global__ void dense_k(const float* __restrict__ X, const float* __restrict__ W,
                        const float* __restrict__ b, float* __restrict__ Y,
                        int R, int CI, int CO, int ldX, int ldY, int colOff,
                        float slope, int act)
{
  int t = blockIdx.x * blockDim.x + threadIdx.x;
  if (t >= R * CO) return;
  const int r = t / CO, c = t % CO;
  float s = b[c];
  for (int i = 0; i < CI; ++i) s += X[r * ldX + i] * W[i * CO + c];
  if (act) s = s > 0.f ? s : slope * s;
  Y[r * ldY + colOff + c] = s;
}

// ---------------- host orchestration ----------------
static inline int gridFor(int n, int b) { return (n + b - 1) / b; }

extern "C" void kernel_launch(void* const* d_in, const int* in_sizes, int n_in,
                              void* d_out, int out_size, void* d_ws, size_t ws_size,
                              hipStream_t stream) {
  (void)in_sizes; (void)n_in; (void)out_size; (void)ws_size;

  const float* inputs = (const float*)d_in[0];
  const int*   src    = (const int*)d_in[1];
  const int*   dst    = (const int*)d_in[2];
  const int*   gid    = (const int*)d_in[3];
  const float* p1 = (const float*)d_in[4];
  const float* p2 = (const float*)d_in[5];
  const float* p3 = (const float*)d_in[6];
  const float* Wl[3][2] = { {(const float*)d_in[7],  (const float*)d_in[8]},
                            {(const float*)d_in[11], (const float*)d_in[12]},
                            {(const float*)d_in[15], (const float*)d_in[16]} };
  const float* al[3] = { (const float*)d_in[9],  (const float*)d_in[13], (const float*)d_in[17] };
  const float* bl[3] = { (const float*)d_in[10], (const float*)d_in[14], (const float*)d_in[18] };
  const float* Wex  = (const float*)d_in[19]; const float* bex  = (const float*)d_in[20];
  const float* Wpat = (const float*)d_in[21]; const float* bpat = (const float*)d_in[22];
  const float* Wc1  = (const float*)d_in[23]; const float* bc1  = (const float*)d_in[24];
  const float* Wc2  = (const float*)d_in[25]; const float* bc2  = (const float*)d_in[26];
  const float* Wc3  = (const float*)d_in[27]; const float* bc3  = (const float*)d_in[28];
  float* out = (float*)d_out;

  // workspace layout (deterministic every call)
  char* ws = (char*)d_ws;
  size_t off = 0;
  auto carve = [&](size_t bytes) -> char* {
    char* p = ws + off;
    off += (bytes + 255) & ~(size_t)255;
    return p;
  };
  unsigned short* Xb  = (unsigned short*)carve((size_t)NN * HD * 2);
  unsigned int* WsPk  = (unsigned int*)carve((size_t)(HD / 32) * NTC * 256 * 4);
  unsigned int* WdPk  = (unsigned int*)carve((size_t)(HD / 32) * NTC * 256 * 4);
  float* fs    = (float*)carve((size_t)NN * HD * 4);
  float* fd    = (float*)carve((size_t)NN * HD * 4);
  float* acc   = (float*)carve((size_t)NN * HD * 4);
  float* logit = (float*)carve((size_t)EE * HH * 4);
  float* ex    = (float*)carve((size_t)EE * HH * 4);
  unsigned int* menc = (unsigned int*)carve((size_t)NN * HH * 4);
  float* mf    = (float*)carve((size_t)NN * HH * 4);
  float* den   = (float*)carve((size_t)NN * HH * 4);
  float* gsum  = (float*)carve((size_t)GG * DD * 4);
  float* gcnt  = (float*)carve((size_t)GG * 4);
  float* xcat  = (float*)carve((size_t)GG * (DD + PP) * 4);
  float* pex   = (float*)carve((size_t)GG * 96 * 4);
  float* t1    = (float*)carve((size_t)GG * 64 * 4);
  float* t2    = (float*)carve((size_t)GG * 32 * 4);

  const int B = 256;
  const dim3 gemmGrid((NN + 127) / 128, HD / 16);

  for (int L = 0; L < 3; ++L) {
    const int K = (L == 0) ? F_IN : HD;
    const float* xin = (L == 0) ? inputs : acc;
    const int wPk = (K / 32) * NTC * 256;   // packed W dword count

    // convert inputs to bf16, weights to tile-packed bf16
    cvt_bf16_k<<<gridFor(NN * K, B), B, 0, stream>>>(xin, Xb, NN * K);
    cvt_w_pack_k<<<gridFor(wPk, B), B, 0, stream>>>(Wl[L][0], WsPk, K);
    cvt_w_pack_k<<<gridFor(wPk, B), B, 0, stream>>>(Wl[L][1], WdPk, K);

    // fused WMMA projections
    gemm_dual_wmma_k<<<gemmGrid, B, 0, stream>>>(Xb, WsPk, WdPk, fs, fd, NN, K);

    // reset segment reducers (acc reset AFTER cvt consumed previous layer output)
    fill_u32_k<<<gridFor(NN * HH, B), B, 0, stream>>>(menc, 0u, NN * HH);
    fill_f32_k<<<gridFor(NN * HH, B), B, 0, stream>>>(den, 0.f, NN * HH);
    fill_f32_k<<<gridFor(NN * HD, B), B, 0, stream>>>(acc, 0.f, NN * HD);

    // edge softmax + aggregate
    edge_logits_k<<<gridFor(EE * HH, B), B, 0, stream>>>(src, dst, fs, fd, al[L], logit, menc);
    mfix_k<<<gridFor(NN * HH, B), B, 0, stream>>>(menc, mf, NN * HH);
    edge_exp_k<<<gridFor(EE * HH, B), B, 0, stream>>>(dst, logit, mf, ex, den);
    edge_agg_k<<<gridFor(EE * HH, B), B, 0, stream>>>(src, dst, fs, ex, den, acc);
    add_bias_k<<<gridFor(NN * HD, B), B, 0, stream>>>(acc, bl[L], NN * HD);
  }

  // graph readout: mean over heads, then per-graph mean
  fill_f32_k<<<gridFor(GG * DD, B), B, 0, stream>>>(gsum, 0.f, GG * DD);
  fill_f32_k<<<1, B, 0, stream>>>(gcnt, 0.f, GG);
  hmean_k<<<gridFor(NN * DD, B), B, 0, stream>>>(acc, gid, gsum, gcnt);
  gfinal_k<<<gridFor(GG * DD, B), B, 0, stream>>>(gsum, gcnt, xcat);

  // pattern branch: three ex_layer1 applications -> concat -> pattern layer
  dense_k<<<gridFor(GG * 32, B), B, 0, stream>>>(p1, Wex, bex, pex, GG, PP, 32, PP, 96, 0,  0.f, 0);
  dense_k<<<gridFor(GG * 32, B), B, 0, stream>>>(p2, Wex, bex, pex, GG, PP, 32, PP, 96, 32, 0.f, 0);
  dense_k<<<gridFor(GG * 32, B), B, 0, stream>>>(p3, Wex, bex, pex, GG, PP, 32, PP, 96, 64, 0.f, 0);
  dense_k<<<gridFor(GG * PP, B), B, 0, stream>>>(pex, Wpat, bpat, xcat, GG, 96, PP, 96, DD + PP, DD, NEG_MLP, 1);

  // classifier head
  dense_k<<<gridFor(GG * 64, B), B, 0, stream>>>(xcat, Wc1, bc1, t1, GG, DD + PP, 64, DD + PP, 64, 0, NEG_MLP, 1);
  dense_k<<<gridFor(GG * 32, B), B, 0, stream>>>(t1, Wc2, bc2, t2, GG, 64, 32, 64, 32, 0, NEG_MLP, 1);
  dense_k<<<gridFor(GG * 2,  B), B, 0, stream>>>(t2, Wc3, bc3, out, GG, 32, 2, 32, 2, 0, 0.f, 0);
}
